// GroupedQueryAttention_66898410602872
// MI455X (gfx1250) — matrix operand order repaired and checked by reading, not compiled, not run on hardware
//
#include <hip/hip_runtime.h>
#include <hip/hip_bf16.h>
#include <math.h>

typedef __attribute__((ext_vector_type(16))) __bf16 v16bf;
typedef __attribute__((ext_vector_type(8)))  float  v8f;
typedef __attribute__((ext_vector_type(4)))  unsigned int v4u;
typedef __attribute__((ext_vector_type(8)))  int  v8i;
typedef __attribute__((ext_vector_type(4)))  int  v4i;

union Frag {
    v16bf v;
    unsigned int u[8];
};

__device__ __forceinline__ unsigned short f2bf(float f) {
    unsigned int u = __float_as_uint(f);
    u += 0x7FFFu + ((u >> 16) & 1u);   // round-to-nearest-even
    return (unsigned short)(u >> 16);
}

// ---------------------------------------------------------------- convert
__global__ void cvt_f32_bf16(const float* __restrict__ in,
                             unsigned short* __restrict__ out, int n) {
    int i = blockIdx.x * blockDim.x + threadIdx.x;
    if (i < n) out[i] = f2bf(in[i]);
}

// Transpose K x N fp32 -> N x K bf16 (weights become N-major for the GEMM B side)
__global__ __launch_bounds__(256) void transpose_f32_bf16(
    const float* __restrict__ in, unsigned short* __restrict__ out, int K, int N) {
    __shared__ float t[32][33];
    int k0 = blockIdx.y * 32, n0 = blockIdx.x * 32;
#pragma unroll
    for (int r = threadIdx.y; r < 32; r += 8)
        t[r][threadIdx.x] = in[(size_t)(k0 + r) * N + n0 + threadIdx.x];
    __syncthreads();
#pragma unroll
    for (int r = threadIdx.y; r < 32; r += 8)
        out[(size_t)(n0 + r) * K + k0 + threadIdx.x] = f2bf(t[threadIdx.x][r]);
}

// ---------------------------------------------------------------- TDM helper
// 2D tile DMA: global (row-major, row length row_stride elems of 2 bytes)
// -> LDS packed rows of tile_k elems.  D# per CDNA5 ISA §8.
__device__ __forceinline__ void tdm_load_2d(unsigned lds_addr, const void* gaddr,
                                            unsigned tile_k, unsigned tile_rows,
                                            unsigned row_stride, unsigned rows_avail)
{
    unsigned long long ga = (unsigned long long)(size_t)gaddr;
    v4u g0;
    g0[0] = 1u;                                            // count=1, user D#
    g0[1] = lds_addr;                                      // lds_addr (bytes)
    g0[2] = (unsigned)(ga & 0xFFFFFFFFu);                  // global_addr[31:0]
    g0[3] = (unsigned)((ga >> 32) & 0x1FFFFFFu) | (2u << 30); // addr[56:32] | type=2
    v8i g1;
    g1[0] = (int)(1u << 16);                               // data_size=1 (2 bytes)
    g1[1] = (int)((row_stride & 0xFFFFu) << 16);           // tensor_dim0[15:0]
    g1[2] = (int)(((row_stride >> 16) & 0xFFFFu) |
                  ((rows_avail & 0xFFFFu) << 16));         // dim0 hi | tensor_dim1 lo
    g1[3] = (int)(((rows_avail >> 16) & 0xFFFFu) |
                  ((tile_k & 0xFFFFu) << 16));             // dim1 hi | tile_dim0
    g1[4] = (int)(tile_rows & 0xFFFFu);                    // tile_dim1 (tile_dim2=0)
    g1[5] = (int)row_stride;                               // tensor_dim0_stride[31:0]
    g1[6] = 0;
    g1[7] = 0;
    v4i gz4 = {0, 0, 0, 0};
    v8i gz8 = {0, 0, 0, 0, 0, 0, 0, 0};
    __builtin_amdgcn_tensor_load_to_lds(g0, g1, gz4, gz4, gz8, 0);
}

// ---------------------------------------------------------------- GEMM
// C(MxN) = A(MxK,bf16,row-major) @ B^T where Bt is (NxK,bf16,row-major).
// Block tile 64x128, 8 waves of 32x32 wave tiles, BK=32, TDM double-buffered.
#define BM 64
#define BN 128
#define BKK 32

// epilogue modes
#define EPI_F32  0
#define EPI_BF16 1
#define EPI_KV_K 2   // f32 -> (B,HKV,S,HD); bf16 copy same layout
#define EPI_KV_V 3   // f32 -> (B,HKV,S,HD); bf16 copy transposed per head (HD,S)

__global__ __launch_bounds__(256) void gemm_bf16_wmma(
    const unsigned short* __restrict__ A, const unsigned short* __restrict__ Bt,
    int M, int N, int K, int mode,
    float* __restrict__ Cf, unsigned short* __restrict__ Cb,
    int Sdim, int HKVdim, int HDdim)
{
    __shared__ unsigned short lA[2][BM * BKK];
    __shared__ unsigned short lB[2][BN * BKK];

    const int tid  = threadIdx.x;
    const int lane = tid & 31;
    const int wave = tid >> 5;
    const int lh   = lane >> 4;
    const int l16  = lane & 15;
    const int bn0  = blockIdx.x * BN;
    const int bm0  = blockIdx.y * BM;
    const int wm   = (wave >> 2) * 32;
    const int wn   = (wave & 3) * 32;

    const unsigned ldsA[2] = { (unsigned)(size_t)(&lA[0][0]), (unsigned)(size_t)(&lA[1][0]) };
    const unsigned ldsB[2] = { (unsigned)(size_t)(&lB[0][0]), (unsigned)(size_t)(&lB[1][0]) };

    v8f c[2][2];
#pragma unroll
    for (int mi = 0; mi < 2; ++mi)
#pragma unroll
        for (int ni = 0; ni < 2; ++ni)
#pragma unroll
            for (int r = 0; r < 8; ++r) c[mi][ni][r] = 0.0f;

    const int nkt = K / BKK;

    // prologue: prefetch tile 0 into buffer 0
    if (wave == 0) {
        tdm_load_2d(ldsA[0], A  + (size_t)bm0 * K, BKK, BM, (unsigned)K, (unsigned)(M - bm0));
        tdm_load_2d(ldsB[0], Bt + (size_t)bn0 * K, BKK, BN, (unsigned)K, (unsigned)(N - bn0));
    }

    for (int kt = 0; kt < nkt; ++kt) {
        const int cur = kt & 1;
        if (wave == 0)
            __builtin_amdgcn_s_wait_tensorcnt(0);
        __syncthreads();           // buffer `cur` ready for everyone

        // prefetch next tile into the other buffer
        if (wave == 0 && (kt + 1) < nkt) {
            const size_t ko = (size_t)(kt + 1) * BKK;
            tdm_load_2d(ldsA[cur ^ 1], A  + (size_t)bm0 * K + ko, BKK, BM, (unsigned)K, (unsigned)(M - bm0));
            tdm_load_2d(ldsB[cur ^ 1], Bt + (size_t)bn0 * K + ko, BKK, BN, (unsigned)K, (unsigned)(N - bn0));
        }

        const unsigned short* pA = &lA[cur][0];
        const unsigned short* pB = &lB[cur][0];

        Frag a[2], b[2];
#pragma unroll
        for (int mi = 0; mi < 2; ++mi) {
            int mrow = wm + mi * 16 + l16;
#pragma unroll
            for (int i = 0; i < 8; ++i) {
                int kk = lh * 16 + 2 * i;
                a[mi].u[i] = *(const unsigned int*)(pA + mrow * BKK + kk);
            }
        }
#pragma unroll
        for (int ni = 0; ni < 2; ++ni) {
            int ncol = wn + ni * 16 + l16;
#pragma unroll
            for (int i = 0; i < 8; ++i) {
                int kk = lh * 16 + 2 * i;
                b[ni].u[i] = *(const unsigned int*)(pB + ncol * BKK + kk);
            }
        }
#pragma unroll
        for (int mi = 0; mi < 2; ++mi)
#pragma unroll
            for (int ni = 0; ni < 2; ++ni)
                c[mi][ni] = __builtin_amdgcn_wmma_f32_16x16x32_bf16(
                    false, a[mi].v, false, b[ni].v, (short)0, c[mi][ni], false, false);
        __syncthreads();           // everyone done reading `cur` before it is reloaded
    }

    // epilogue
#pragma unroll
    for (int mi = 0; mi < 2; ++mi)
#pragma unroll
        for (int ni = 0; ni < 2; ++ni)
#pragma unroll
            for (int r = 0; r < 8; ++r) {
                int gm = bm0 + wm + mi * 16 + r + lh * 8;
                int gn = bn0 + wn + ni * 16 + l16;
                float val = c[mi][ni][r];
                if (mode == EPI_F32) {
                    Cf[(size_t)gm * N + gn] = val;
                } else if (mode == EPI_BF16) {
                    Cb[(size_t)gm * N + gn] = f2bf(val);
                } else {
                    int bb = gm / Sdim, s = gm % Sdim;
                    int h  = gn / HDdim, hd = gn % HDdim;
                    size_t o = (((size_t)bb * HKVdim + h) * Sdim + s) * HDdim + hd;
                    Cf[o] = val;
                    if (mode == EPI_KV_K) {
                        Cb[o] = f2bf(val);
                    } else { // EPI_KV_V: per-head transposed (hd, s)
                        size_t ot = (((size_t)bb * HKVdim + h) * HDdim + hd) * Sdim + s;
                        Cb[ot] = f2bf(val);
                    }
                }
            }
}

// ---------------------------------------------------------------- attention
#define AS  2048
#define AD  2048
#define AHQ 32
#define AHKV 8
#define AHD 64
#define AG  4

__global__ __launch_bounds__(256) void gqa_attn_wmma(
    const unsigned short* __restrict__ Q,   // (B*S, D) bf16, head slice h*64
    const unsigned short* __restrict__ Kb,  // (B,HKV,S,HD) bf16 row-major per head
    const unsigned short* __restrict__ Vt,  // (B,HKV,HD,S) bf16 transposed per head
    unsigned short* __restrict__ AO)        // (B*S, D) bf16
{
    __shared__ unsigned short ldsP[8 * 16 * 32];

    const int lane = threadIdx.x & 31;
    const int wv   = threadIdx.x >> 5;
    const int lh   = lane >> 4;
    const int l16  = lane & 15;

    const int gw   = blockIdx.x * 8 + wv;
    const int nqt  = AS / 16;
    const int qt   = gw % nqt;
    const int rest = gw / nqt;
    const int h    = rest % AHQ;
    const int b    = rest / AHQ;
    const int s0   = qt * 16;
    const int kvh  = h / AG;

    const unsigned short* qp = Q  + (size_t)b * AS * AD + (size_t)h * AHD;
    const unsigned short* kp = Kb + ((size_t)(b * AHKV + kvh)) * AS * AHD;
    const unsigned short* vp = Vt + ((size_t)(b * AHKV + kvh)) * AHD * AS;
    unsigned short* pbuf = ldsP + wv * 16 * 32;

    Frag qa[2];
#pragma unroll
    for (int kh = 0; kh < 2; ++kh)
#pragma unroll
        for (int i = 0; i < 8; ++i) {
            int kk = lh * 16 + 2 * i + kh * 32;
            qa[kh].u[i] = *(const unsigned int*)(qp + (size_t)(s0 + l16) * AD + kk);
        }

    v8f acc[4];
#pragma unroll
    for (int j = 0; j < 4; ++j)
#pragma unroll
        for (int r = 0; r < 8; ++r) acc[j][r] = 0.0f;
    float rm[8], rl[8];
#pragma unroll
    for (int r = 0; r < 8; ++r) { rm[r] = -INFINITY; rl[r] = 0.0f; }

    for (int t0 = 0; t0 <= s0 + 15; t0 += 32) {
        v8f sc[2];
#pragma unroll
        for (int ct = 0; ct < 2; ++ct) {
            Frag kb0, kb1;
            int col = t0 + ct * 16 + l16;
#pragma unroll
            for (int i = 0; i < 8; ++i) {
                int kk = lh * 16 + 2 * i;
                kb0.u[i] = *(const unsigned int*)(kp + (size_t)col * AHD + kk);
                kb1.u[i] = *(const unsigned int*)(kp + (size_t)col * AHD + kk + 32);
            }
            v8f z;
#pragma unroll
            for (int r = 0; r < 8; ++r) z[r] = 0.0f;
            z = __builtin_amdgcn_wmma_f32_16x16x32_bf16(false, qa[0].v, false, kb0.v, (short)0, z, false, false);
            z = __builtin_amdgcn_wmma_f32_16x16x32_bf16(false, qa[1].v, false, kb1.v, (short)0, z, false, false);
            sc[ct] = z;
        }

        float p[2][8];
#pragma unroll
        for (int ct = 0; ct < 2; ++ct) {
            int ng = t0 + ct * 16 + l16;
#pragma unroll
            for (int r = 0; r < 8; ++r) {
                int mg = s0 + r + lh * 8;
                float v = sc[ct][r] * 0.125f;       // 1/sqrt(64)
                p[ct][r] = (ng > mg) ? -INFINITY : v;
            }
        }

#pragma unroll
        for (int r = 0; r < 8; ++r) {
            float loc = fmaxf(p[0][r], p[1][r]);
            loc = fmaxf(loc, __shfl_xor(loc, 1, 32));
            loc = fmaxf(loc, __shfl_xor(loc, 2, 32));
            loc = fmaxf(loc, __shfl_xor(loc, 4, 32));
            loc = fmaxf(loc, __shfl_xor(loc, 8, 32));
            float mnew = fmaxf(rm[r], loc);
            float corr = __expf(rm[r] - mnew);
            float e0 = __expf(p[0][r] - mnew);
            float e1 = __expf(p[1][r] - mnew);
            float sum = e0 + e1;
            sum += __shfl_xor(sum, 1, 32);
            sum += __shfl_xor(sum, 2, 32);
            sum += __shfl_xor(sum, 4, 32);
            sum += __shfl_xor(sum, 8, 32);
            rl[r] = rl[r] * corr + sum;
            rm[r] = mnew;
#pragma unroll
            for (int j = 0; j < 4; ++j) acc[j][r] *= corr;
            int m = r + lh * 8;
            pbuf[m * 32 + l16]      = f2bf(e0);
            pbuf[m * 32 + 16 + l16] = f2bf(e1);
        }

        Frag pa;
#pragma unroll
        for (int i = 0; i < 8; ++i) {
            int kk = lh * 16 + 2 * i;
            pa.u[i] = *(const unsigned int*)(pbuf + l16 * 32 + kk);
        }
#pragma unroll
        for (int j = 0; j < 4; ++j) {
            Frag vb;
#pragma unroll
            for (int i = 0; i < 8; ++i) {
                int kk = lh * 16 + 2 * i;
                vb.u[i] = *(const unsigned int*)(vp + (size_t)(j * 16 + l16) * AS + t0 + kk);
            }
            acc[j] = __builtin_amdgcn_wmma_f32_16x16x32_bf16(false, pa.v, false, vb.v, (short)0, acc[j], false, false);
        }
    }

#pragma unroll
    for (int j = 0; j < 4; ++j)
#pragma unroll
        for (int r = 0; r < 8; ++r) {
            int m = r + lh * 8;
            float o = acc[j][r] / rl[r];
            AO[(size_t)(b * AS + s0 + m) * AD + h * AHD + j * 16 + l16] = f2bf(o);
        }
}

// ---------------------------------------------------------------- launch
extern "C" void kernel_launch(void* const* d_in, const int* in_sizes, int n_in,
                              void* d_out, int out_size, void* d_ws, size_t ws_size,
                              hipStream_t stream) {
    const float* x  = (const float*)d_in[0];
    const float* Wq = (const float*)d_in[1];
    const float* Wk = (const float*)d_in[2];
    const float* Wv = (const float*)d_in[3];
    const float* Wo = (const float*)d_in[4];
    // d_in[5] (mask) unused: causal mask computed analytically.
    float* out = (float*)d_out;

    const int Bd = 2, S = 2048, D = 2048, HQ = 32, HKV = 8, HD = 64;
    const int KV = HKV * HD;                      // 512
    const size_t nX  = (size_t)Bd * S * D;        // 8388608
    const size_t nWq = (size_t)D * D;             // 4194304
    const size_t nWk = (size_t)D * KV;            // 1048576
    const size_t nKV = (size_t)Bd * HKV * S * HD; // 2097152

    unsigned char* ws = (unsigned char*)d_ws;
    size_t off = 0;
    auto walloc = [&](size_t bytes) -> void* {
        void* p = ws + off;
        off = (off + bytes + 255) & ~(size_t)255;
        return p;
    };
    unsigned short* xb  = (unsigned short*)walloc(nX * 2);
    unsigned short* wqb = (unsigned short*)walloc(nWq * 2);  // transposed D x D
    unsigned short* wkb = (unsigned short*)walloc(nWk * 2);  // transposed KV x D
    unsigned short* wvb = (unsigned short*)walloc(nWk * 2);  // transposed KV x D
    unsigned short* wob = (unsigned short*)walloc(nWq * 2);  // transposed D x D
    unsigned short* qb  = (unsigned short*)walloc(nX * 2);
    unsigned short* kbb = (unsigned short*)walloc(nKV * 2);
    unsigned short* vtb = (unsigned short*)walloc(nKV * 2);
    unsigned short* aob = (unsigned short*)walloc(nX * 2);

    float* outO = out;
    float* outK = out + nX;
    float* outV = out + nX + nKV;

    cvt_f32_bf16<<<(int)(nX / 256), 256, 0, stream>>>(x, xb, (int)nX);
    // weights: K x N fp32 -> N x K bf16
    transpose_f32_bf16<<<dim3(D / 32, D / 32),  dim3(32, 8), 0, stream>>>(Wq, wqb, D, D);
    transpose_f32_bf16<<<dim3(KV / 32, D / 32), dim3(32, 8), 0, stream>>>(Wk, wkb, D, KV);
    transpose_f32_bf16<<<dim3(KV / 32, D / 32), dim3(32, 8), 0, stream>>>(Wv, wvb, D, KV);
    transpose_f32_bf16<<<dim3(D / 32, D / 32),  dim3(32, 8), 0, stream>>>(Wo, wob, D, D);

    const int M = Bd * S;
    // Q = x @ W_q  (bf16 out)
    gemm_bf16_wmma<<<dim3(D / BN, M / BM), 256, 0, stream>>>(
        xb, wqb, M, D, D, EPI_BF16, nullptr, qb, S, HKV, HD);
    // K = x @ W_k  (f32 -> d_out transposed heads; bf16 row-major per head)
    gemm_bf16_wmma<<<dim3(KV / BN, M / BM), 256, 0, stream>>>(
        xb, wkb, M, KV, D, EPI_KV_K, outK, kbb, S, HKV, HD);
    // V = x @ W_v  (f32 -> d_out transposed heads; bf16 (hd,s) per head)
    gemm_bf16_wmma<<<dim3(KV / BN, M / BM), 256, 0, stream>>>(
        xb, wvb, M, KV, D, EPI_KV_V, outV, vtb, S, HKV, HD);
    // flash attention
    gqa_attn_wmma<<<(Bd * HQ * (S / 16)) / 8, 256, 0, stream>>>(qb, kbb, vtb, aob);
    // out = attn_out @ W_o (f32)
    gemm_bf16_wmma<<<dim3(D / BN, M / BM), 256, 0, stream>>>(
        aob, wob, M, D, D, EPI_F32, outO, nullptr, S, HKV, HD);
}
